// GRU_D_layer_72756745994364
// MI455X (gfx1250) — matrix operand order repaired and checked by reading, compiled, not run on hardware
//
#include <hip/hip_runtime.h>
#include <hip/hip_bf16.h>

// GRU-D layer for MI455X (gfx1250, wave32, WMMA).
// B=64, T=512, D=256, H=1024.
//
// Phase 1: pack weights transposed to bf16; one WMMA GEMM kernel computes all
//          four input projections (decay/z/r/h) and stores gamma, xz, xr, xh
//          as bf16 in [T,B,H] layout.
// Phase 2: 512 serialized steps; per step two WMMA kernels:
//          gates:  z = sigmoid(xz + h_dec @ Uz), r = sigmoid(xr + h_dec @ Ur)
//                  -> writes r*h_dec in WMMA A-fragment order
//          cand:   stages r*h_dec via async global->LDS copy (ASYNCcnt),
//                  hp = tanh(xh + (r*h_dec) @ Uh); h = (1-z)*h_dec + z*hp

#define BB 64
#define TT 512
#define DD 256
#define HH 1024

typedef __attribute__((ext_vector_type(16))) __bf16 v16bf;
typedef __attribute__((ext_vector_type(8)))  __bf16 v8bf;
typedef __attribute__((ext_vector_type(8)))  float  v8f;

// 16-byte int vector type matching the async-LDS builtin's parameter type.
typedef int v4i_ __attribute__((__vector_size__(16)));
typedef __attribute__((address_space(1))) v4i_ gas_v4i;  // global
typedef __attribute__((address_space(3))) v4i_ las_v4i;  // LDS

// WMMA 16-bit A-matrix (16x32) fragment layout (contiguous K order):
// lane = m + 16*g ; element e (0..15): k = 16g + e
__device__ __forceinline__ int a_klocal(int e, int g) {
  return 16 * g + e;
}

// Inverse map: (row-in-tile mr, K index k) -> linear slot in a 16xK fragment
// buffer laid out [kb][lane][e] (512 bf16 per 32-wide K block).
__device__ __forceinline__ int a_slot(int mr, int k) {
  int kb = k >> 5, kl = k & 31;
  int g = kl >> 4;
  int e = kl & 15;
  return (kb << 9) + ((mr + (g << 4)) << 4) + e;
}

__device__ __forceinline__ float sigmoidf_(float xv) {
  return 1.0f / (1.0f + __expf(-xv));
}

// ---------------------------------------------------------------------------
// Pack f32 [K][N] -> bf16 [N][K] (K contiguous per column => contiguous 32B
// B-fragment loads in the GEMM kernels).
__global__ void __launch_bounds__(256) transpose_pack_kernel(
    const float* __restrict__ src, __bf16* __restrict__ dst, int K, int N) {
  int idx = blockIdx.x * 256 + threadIdx.x;
  if (idx >= K * N) return;
  int n = idx / K;
  int k = idx - n * K;
  dst[idx] = (__bf16)src[(size_t)k * N + n];
}

__global__ void __launch_bounds__(256) zero_f32_kernel(float* __restrict__ p, int n) {
  int idx = blockIdx.x * 256 + threadIdx.x;
  if (idx < n) p[idx] = 0.0f;
}

// ---------------------------------------------------------------------------
// Phase 1: fused 4-way projection GEMM.
// Grid: 2048 M-tiles (16 rows of B*T) x 8 N-groups (128 cols of H).
__global__ void __launch_bounds__(256) grud_proj_kernel(
    const float* __restrict__ x,
    const __bf16* __restrict__ WdT, const __bf16* __restrict__ WzT,
    const __bf16* __restrict__ WrT, const __bf16* __restrict__ WhT,
    const float* __restrict__ bd, const float* __restrict__ bz,
    const float* __restrict__ br, const float* __restrict__ bh,
    __bf16* __restrict__ gamma, __bf16* __restrict__ xz,
    __bf16* __restrict__ xr, __bf16* __restrict__ xh) {
  __shared__ __align__(32) __bf16 aFrag[8 * 512];  // 16 rows x 256 K
  const int mt = blockIdx.x & 2047;
  const int ng = blockIdx.x >> 11;
  const int m0 = mt << 4;

  for (int s = threadIdx.x; s < 16 * DD; s += 256) {
    int kb = s >> 9, fs = s & 511;
    int ln = fs >> 4, e = fs & 15;
    int g = ln >> 4, mr = ln & 15;
    int k = (kb << 5) + a_klocal(e, g);
    aFrag[s] = (__bf16)x[(size_t)(m0 + mr) * DD + k];
  }
  __syncthreads();

  const int wave = threadIdx.x >> 5;
  const int lane = threadIdx.x & 31;
  const int n0 = (ng << 7) + (wave << 4);
  const int ncol = lane & 15;
  const int g = lane >> 4;

  v8f acc0 = {}, acc1 = {}, acc2 = {}, acc3 = {};
  for (int kb = 0; kb < DD / 32; ++kb) {
    v16bf a = *(const v16bf*)&aFrag[(kb << 9) + (lane << 4)];
    size_t boff = ((size_t)(n0 + ncol) << 8) + (kb << 5) + (g << 4);
    v16bf b0 = *(const v16bf*)(WdT + boff);
    v16bf b1 = *(const v16bf*)(WzT + boff);
    v16bf b2 = *(const v16bf*)(WrT + boff);
    v16bf b3 = *(const v16bf*)(WhT + boff);
    acc0 = __builtin_amdgcn_wmma_f32_16x16x32_bf16(false, a, false, b0, (short)0, acc0, false, false);
    acc1 = __builtin_amdgcn_wmma_f32_16x16x32_bf16(false, a, false, b1, (short)0, acc1, false, false);
    acc2 = __builtin_amdgcn_wmma_f32_16x16x32_bf16(false, a, false, b2, (short)0, acc2, false, false);
    acc3 = __builtin_amdgcn_wmma_f32_16x16x32_bf16(false, a, false, b3, (short)0, acc3, false, false);
  }

  // C/D layout (ISA 7.12.2): VGPR i, lane L: n = L&15, m = i + 8*(L>>4)
  for (int i = 0; i < 8; ++i) {
    int m = i + ((lane >> 4) << 3);
    int grow = m0 + m;               // global row in [0, B*T)
    int brow = grow >> 9;            // batch index (T = 512)
    int trow = grow & 511;           // time index
    int h = n0 + ncol;
    size_t oidx = (size_t)trow * (BB * HH) + (size_t)brow * HH + h;
    float dv = acc0[i] + bd[h];
    gamma[oidx] = (__bf16)__expf(-fmaxf(dv, 0.0f));
    xz[oidx] = (__bf16)(acc1[i] + bz[h]);
    xr[oidx] = (__bf16)(acc2[i] + br[h]);
    xh[oidx] = (__bf16)(acc3[i] + bh[h]);
  }
}

// ---------------------------------------------------------------------------
// Phase 2a: gates. Grid: 4 M-tiles x 8 N-groups = 32 workgroups.
// Stages h_dec = gamma_t * h_{t-1} (16 x 1024) into LDS in A-fragment order,
// then each wave runs two 32-step WMMA K-chains (Uz and Ur share A fragment).
// Epilogue writes r*h_dec directly in A-fragment order for the cand kernel.
__global__ void __launch_bounds__(256) grud_gates_kernel(
    const __bf16* __restrict__ gamma, const __bf16* __restrict__ xz,
    const __bf16* __restrict__ xr,
    const __bf16* __restrict__ UzT, const __bf16* __restrict__ UrT,
    const float* __restrict__ hprev,
    float* __restrict__ z_out, __bf16* __restrict__ rh_frag, int t) {
  __shared__ __align__(32) __bf16 aFrag[32 * 512];  // 16 x 1024, 32 KB
  const int mt = blockIdx.x & 3;
  const int ng = blockIdx.x >> 2;
  const int m0 = mt << 4;
  const size_t tbase = (size_t)t * (BB * HH);

  for (int s = threadIdx.x; s < 16 * HH; s += 256) {
    int kb = s >> 9, fs = s & 511;
    int ln = fs >> 4, e = fs & 15;
    int g = ln >> 4, mr = ln & 15;
    int k = (kb << 5) + a_klocal(e, g);
    size_t bi = (size_t)(m0 + mr) * HH + k;
    aFrag[s] = (__bf16)((float)gamma[tbase + bi] * hprev[bi]);
  }
  __syncthreads();

  const int wave = threadIdx.x >> 5;
  const int lane = threadIdx.x & 31;
  const int n0 = (ng << 7) + (wave << 4);
  const int ncol = lane & 15;
  const int g = lane >> 4;

  v8f cz = {}, cr = {};
  for (int kb = 0; kb < HH / 32; ++kb) {
    v16bf a = *(const v16bf*)&aFrag[(kb << 9) + (lane << 4)];
    size_t boff = ((size_t)(n0 + ncol) << 10) + (kb << 5) + (g << 4);
    v16bf fz = *(const v16bf*)(UzT + boff);
    v16bf fr = *(const v16bf*)(UrT + boff);
    cz = __builtin_amdgcn_wmma_f32_16x16x32_bf16(false, a, false, fz, (short)0, cz, false, false);
    cr = __builtin_amdgcn_wmma_f32_16x16x32_bf16(false, a, false, fr, (short)0, cr, false, false);
  }

  for (int i = 0; i < 8; ++i) {
    int m = i + ((lane >> 4) << 3);
    int brow = m0 + m;
    int h = n0 + ncol;
    size_t bi = (size_t)brow * HH + h;
    size_t ti = tbase + bi;
    float zv = sigmoidf_((float)xz[ti] + cz[i]);
    float rv = sigmoidf_((float)xr[ti] + cr[i]);
    float hd = (float)gamma[ti] * hprev[bi];
    z_out[bi] = zv;
    // A-fragment-order store: cand kernel stages this with an async copy.
    rh_frag[((size_t)(brow >> 4) << 14) + a_slot(brow & 15, h)] = (__bf16)(rv * hd);
  }
}

// ---------------------------------------------------------------------------
// Phase 2b: candidate + blend. A operand arrives pre-swizzled, so staging is
// a pure contiguous 32 KB copy -> use GLOBAL_LOAD_ASYNC_TO_LDS_B128 when the
// toolchain exposes it (tracked by ASYNCcnt), else a vector-copy fallback.
__global__ void __launch_bounds__(256) grud_cand_kernel(
    const __bf16* __restrict__ xh, const __bf16* __restrict__ UhT,
    const __bf16* __restrict__ rh_frag, const float* __restrict__ z_in,
    const __bf16* __restrict__ gamma, const float* __restrict__ hprev,
    float* __restrict__ hnext, float* __restrict__ out, int t) {
  __shared__ __align__(32) __bf16 aFrag[32 * 512];
  const int mt = blockIdx.x & 3;
  const int ng = blockIdx.x >> 2;
  const int m0 = mt << 4;
  const size_t tbase = (size_t)t * (BB * HH);
  const __bf16* src = rh_frag + ((size_t)mt << 14);

#if defined(__has_builtin) && __has_builtin(__builtin_amdgcn_global_load_async_to_lds_b128)
  // 16384 bf16 = 2048 x 16B chunks; 256 threads x 8 iterations.
  for (int it = 0; it < 8; ++it) {
    int s = ((it << 8) + (int)threadIdx.x) << 3;  // element index, 16B aligned
    __builtin_amdgcn_global_load_async_to_lds_b128(
        (gas_v4i*)(src + s), (las_v4i*)&aFrag[s], 0, 0);
  }
#if __has_builtin(__builtin_amdgcn_s_wait_asynccnt)
  __builtin_amdgcn_s_wait_asynccnt(0);
#else
  asm volatile("s_wait_asynccnt 0" ::: "memory");
#endif
#else
  for (int s8 = threadIdx.x; s8 < (16 * HH) / 8; s8 += 256) {
    *(v8bf*)&aFrag[s8 << 3] = *(const v8bf*)(src + (s8 << 3));
  }
#endif
  __syncthreads();

  const int wave = threadIdx.x >> 5;
  const int lane = threadIdx.x & 31;
  const int n0 = (ng << 7) + (wave << 4);
  const int ncol = lane & 15;
  const int g = lane >> 4;

  v8f ch = {};
  for (int kb = 0; kb < HH / 32; ++kb) {
    v16bf a = *(const v16bf*)&aFrag[(kb << 9) + (lane << 4)];
    size_t boff = ((size_t)(n0 + ncol) << 10) + (kb << 5) + (g << 4);
    v16bf fh = *(const v16bf*)(UhT + boff);
    ch = __builtin_amdgcn_wmma_f32_16x16x32_bf16(false, a, false, fh, (short)0, ch, false, false);
  }

  for (int i = 0; i < 8; ++i) {
    int m = i + ((lane >> 4) << 3);
    int brow = m0 + m;
    int h = n0 + ncol;
    size_t bi = (size_t)brow * HH + h;
    size_t ti = tbase + bi;
    float hp = tanhf((float)xh[ti] + ch[i]);
    float zv = z_in[bi];
    float hd = (float)gamma[ti] * hprev[bi];
    float ht = (1.0f - zv) * hd + zv * hp;
    hnext[bi] = ht;
    out[(size_t)brow * (TT * HH) + (size_t)t * HH + h] = ht;
  }
}

// ---------------------------------------------------------------------------
extern "C" void kernel_launch(void* const* d_in, const int* in_sizes, int n_in,
                              void* d_out, int out_size, void* d_ws, size_t ws_size,
                              hipStream_t stream) {
  (void)in_sizes; (void)n_in; (void)out_size; (void)ws_size;
  const float* x  = (const float*)d_in[0];
  const float* Wd = (const float*)d_in[1];
  const float* bd = (const float*)d_in[2];
  const float* Wz = (const float*)d_in[3];
  const float* Wr = (const float*)d_in[4];
  const float* Wh = (const float*)d_in[5];
  const float* Uz = (const float*)d_in[6];
  const float* Ur = (const float*)d_in[7];
  const float* Uh = (const float*)d_in[8];
  const float* bz = (const float*)d_in[9];
  const float* br = (const float*)d_in[10];
  const float* bh = (const float*)d_in[11];
  float* out = (float*)d_out;

  char* ws = (char*)d_ws;
  size_t off = 0;
  auto carve = [&](size_t bytes) -> char* {
    char* p = ws + off;
    off = (off + bytes + 255) & ~(size_t)255;
    return p;
  };
  const size_t tbh = (size_t)TT * BB * HH;
  __bf16* g_gamma = (__bf16*)carve(tbh * 2);
  __bf16* g_xz    = (__bf16*)carve(tbh * 2);
  __bf16* g_xr    = (__bf16*)carve(tbh * 2);
  __bf16* g_xh    = (__bf16*)carve(tbh * 2);
  __bf16* WT[4];
  for (int i = 0; i < 4; ++i) WT[i] = (__bf16*)carve((size_t)HH * DD * 2);
  __bf16* UT[3];
  for (int i = 0; i < 3; ++i) UT[i] = (__bf16*)carve((size_t)HH * HH * 2);
  float* hb[2];
  hb[0] = (float*)carve((size_t)BB * HH * 4);
  hb[1] = (float*)carve((size_t)BB * HH * 4);
  float*  g_z  = (float*)carve((size_t)BB * HH * 4);
  __bf16* g_rh = (__bf16*)carve((size_t)BB * HH * 2);  // fragment-order r*h_dec

  // Pack all weights transposed + bf16 (U matrices: 6 MB total, L2-resident).
  const float* Ws[4] = {Wd, Wz, Wr, Wh};
  for (int i = 0; i < 4; ++i)
    transpose_pack_kernel<<<(DD * HH + 255) / 256, 256, 0, stream>>>(Ws[i], WT[i], DD, HH);
  const float* Us[3] = {Uz, Ur, Uh};
  for (int i = 0; i < 3; ++i)
    transpose_pack_kernel<<<(HH * HH + 255) / 256, 256, 0, stream>>>(Us[i], UT[i], HH, HH);

  zero_f32_kernel<<<(BB * HH + 255) / 256, 256, 0, stream>>>(hb[0], BB * HH);

  // Phase 1: batched projections (68.7 GFLOP, bf16 WMMA).
  grud_proj_kernel<<<2048 * 8, 256, 0, stream>>>(x, WT[0], WT[1], WT[2], WT[3],
                                                 bd, bz, br, bh,
                                                 g_gamma, g_xz, g_xr, g_xh);

  // Phase 2: serialized scan (512 steps x 3 GEMMs of 64x1024x1024).
  for (int t = 0; t < TT; ++t) {
    float* hp = hb[t & 1];
    float* hn = hb[(t + 1) & 1];
    grud_gates_kernel<<<32, 256, 0, stream>>>(g_gamma, g_xz, g_xr, UT[0], UT[1],
                                              hp, g_z, g_rh, t);
    grud_cand_kernel<<<32, 256, 0, stream>>>(g_xh, UT[2], g_rh, g_z, g_gamma,
                                             hp, hn, out, t);
  }
}
